// Mamba_Block_59631325938214
// MI455X (gfx1250) — compile-verified, matrix operand order checked by reading, not run on hardware
//
#include <hip/hip_runtime.h>
#include <hip/hip_bf16.h>

// ---------------------------------------------------------------------------
// Mamba block (DEPTH=5) for gfx1250: bf16 WMMA GEMMs + scalar selective scan.
// B=4, L=1024, D_MODEL=512, D_INNER=1024, D_STATE=16, DT_RANK=32, CONV=4.
// GEMM: 128 threads / 4 waves, 128(M)x64(N) block tile, wave = 32x64
// (8 x v_wmma_f32_16x16x32_bf16 per 32-wide K step); A tile staged via
// GLOBAL_LOAD_ASYNC_TO_LDS_B128 (ASYNCcnt) when the builtin is available.
// ---------------------------------------------------------------------------

#define BSZ     4
#define SEQL    1024
#define DM      512
#define DI      1024
#define MROWS   (BSZ * SEQL)      // 4096
#define DSTATE  16
#define DTRANK  32
#define DEPTH   5

typedef __bf16 bf16_t;
typedef __attribute__((ext_vector_type(16))) __bf16 v16bf;
typedef __attribute__((ext_vector_type(8)))  float  v8f;
typedef __attribute__((ext_vector_type(4)))  int    v4i;
typedef __attribute__((address_space(1))) v4i* gptr_v4i;   // global int4*
typedef __attribute__((address_space(3))) v4i* lptr_v4i;   // LDS int4*

#if __has_builtin(__builtin_amdgcn_global_load_async_to_lds_b128)
#define HAVE_ASYNC_LDS 1
#else
#define HAVE_ASYNC_LDS 0
#endif

union FragAB { v16bf v; uint4 u[2]; };
union FragC  { v8f   v; float f[8]; };

// round-to-nearest-even f32 -> bf16 without relying on __bf16 arithmetic
static __device__ __forceinline__ bf16_t f2bf(float x) {
  union { float f; unsigned u; } a; a.f = x;
  unsigned r = a.u + 0x7FFFu + ((a.u >> 16) & 1u);
  unsigned short h = (unsigned short)(r >> 16);
  return __builtin_bit_cast(bf16_t, h);
}

static __device__ __forceinline__ float siluf(float x) {
  return x / (1.0f + __expf(-x));
}
static __device__ __forceinline__ float softplusf(float x) {
  return (x > 20.0f) ? x : log1pf(__expf(x));
}
static __device__ __forceinline__ float geluf(float x) {
  return 0.5f * x * (1.0f + erff(x * 0.70710678118654752f));
}

// Load a 16x32 bf16 fragment (A or B operand) from an LDS tile with row
// stride `stride` (in elements). Layout per ISA 7.12.2: lane (g=lane>>4,
// r=lane&15) holds row r, K = 8g..8g+7 and 16+8g..16+8g+7.
static __device__ __forceinline__ v16bf load_frag(const bf16_t* base, int stride, int lane) {
  FragAB fr;
  const int g = lane >> 4, r = lane & 15;
  const bf16_t* p = base + r * stride + 8 * g;
  fr.u[0] = *(const uint4*)(p);
  fr.u[1] = *(const uint4*)(p + 16);
  return fr.v;
}

// Epilogue flag bits
#define EPI_BIAS     1
#define EPI_SOFTPLUS 2
#define EPI_GELU     4
#define EPI_RESID    8
#define EPI_BF16OUT  16

// C[M,N] = A[M,K] (bf16, row-major) @ W[N,K]^T (f32, row-major, converted to
// bf16 on staging). Requires M % 128 == 0, N % 64 == 0, K % 32 == 0.
template <int EPI>
__global__ __launch_bounds__(128) void gemm_bf16_wmma(
    const bf16_t* __restrict__ A, const float* __restrict__ W,
    float* __restrict__ outF, bf16_t* __restrict__ outB,
    const float* __restrict__ bias, const float* __restrict__ resid,
    int N, int K) {
  __shared__ bf16_t As[128 * 40];  // padded stride 40 (80B, 16B-aligned rows)
  __shared__ bf16_t Bs[64 * 40];

  const int n0   = blockIdx.x * 64;
  const int m0   = blockIdx.y * 128;
  const int tid  = threadIdx.x;
  const int wave = tid >> 5;
  const int lane = tid & 31;
  const int wrow = tid >> 1;     // 0..63   (W tile row)
  const int half = tid & 1;      // 16-element half of the 32-wide K slab

  v8f acc[2][4];
#pragma unroll
  for (int mt = 0; mt < 2; ++mt)
#pragma unroll
    for (int nt = 0; nt < 4; ++nt)
#pragma unroll
      for (int j = 0; j < 8; ++j) acc[mt][nt][j] = 0.0f;

  for (int k0 = 0; k0 < K; k0 += 32) {
    // ---- Stage A tile: 128 rows x 32 k, one full row (64B) per thread ----
    {
      const bf16_t* srcA = A + (size_t)(m0 + tid) * K + k0;
      bf16_t* dstA = As + tid * 40;
#if HAVE_ASYNC_LDS
#pragma unroll
      for (int q = 0; q < 4; ++q)
        __builtin_amdgcn_global_load_async_to_lds_b128(
            (gptr_v4i)(srcA + q * 8), (lptr_v4i)(dstA + q * 8), 0, 0);
#else
#pragma unroll
      for (int q = 0; q < 4; ++q) {
        uint4 a = ((const uint4*)srcA)[q];
        ((uint4*)dstA)[q] = a;
      }
#endif
    }
    // ---- Stage W tile: 64 rows x 32 k with f32 -> bf16 conversion ----
    {
      const float* srcW = W + (size_t)(n0 + wrow) * K + k0 + half * 16;
      bf16_t* dstW = Bs + wrow * 40 + half * 16;
#pragma unroll
      for (int j = 0; j < 16; ++j) dstW[j] = f2bf(srcW[j]);
    }
    // ---- Prefetch next K slab while this one is consumed ----
    if (k0 + 32 < K) {
      __builtin_prefetch(A + (size_t)(m0 + tid) * K + k0 + 32, 0, 1);
      __builtin_prefetch(W + (size_t)(n0 + wrow) * K + k0 + 32 + half * 16, 0, 1);
    }
#if HAVE_ASYNC_LDS
#if __has_builtin(__builtin_amdgcn_s_wait_asynccnt)
    __builtin_amdgcn_s_wait_asynccnt(0);
#else
    asm volatile("s_wait_asynccnt 0x0" ::: "memory");
#endif
#endif
    __syncthreads();

    v16bf af0 = load_frag(As + (wave * 32 + 0)  * 40, 40, lane);
    v16bf af1 = load_frag(As + (wave * 32 + 16) * 40, 40, lane);
#pragma unroll
    for (int nt = 0; nt < 4; ++nt) {
      v16bf bfrag = load_frag(Bs + nt * 16 * 40, 40, lane);
      acc[0][nt] = __builtin_amdgcn_wmma_f32_16x16x32_bf16(
          false, af0, false, bfrag, (short)0, acc[0][nt], false, false);
      acc[1][nt] = __builtin_amdgcn_wmma_f32_16x16x32_bf16(
          false, af1, false, bfrag, (short)0, acc[1][nt], false, false);
    }
    __syncthreads();
  }

  // Epilogue: lane -> (n = lane&15, m rows j + 8*(lane>>4))
  const int g = lane >> 4, nl = lane & 15;
#pragma unroll
  for (int mt = 0; mt < 2; ++mt) {
#pragma unroll
    for (int nt = 0; nt < 4; ++nt) {
      const int n = n0 + nt * 16 + nl;
      FragC c; c.v = acc[mt][nt];
#pragma unroll
      for (int j = 0; j < 8; ++j) {
        const int m = m0 + wave * 32 + mt * 16 + g * 8 + j;
        float v = c.f[j];
        if (EPI & EPI_BIAS)     v += bias[n];
        if (EPI & EPI_SOFTPLUS) v = softplusf(v);
        if (EPI & EPI_GELU)     v = geluf(v);
        if (EPI & EPI_RESID)    v += resid[(size_t)m * N + n];
        if (EPI & EPI_BF16OUT)  outB[(size_t)m * N + n] = f2bf(v);
        else                    outF[(size_t)m * N + n] = v;
      }
    }
  }
}

// LayerNorm over D=512, one block (128 threads) per row; bf16 output.
__global__ __launch_bounds__(128) void ln_to_bf16(
    const float* __restrict__ x, const float* __restrict__ gam,
    const float* __restrict__ bet, bf16_t* __restrict__ out) {
  __shared__ float s1[128], s2[128];
  const int t = threadIdx.x;
  const float* xr = x + (size_t)blockIdx.x * DM;
  float v[4], sum = 0.0f, sq = 0.0f;
#pragma unroll
  for (int i = 0; i < 4; ++i) {
    float val = xr[t + i * 128];
    v[i] = val; sum += val; sq += val * val;
  }
  s1[t] = sum; s2[t] = sq;
  __syncthreads();
  for (int off = 64; off > 0; off >>= 1) {
    if (t < off) { s1[t] += s1[t + off]; s2[t] += s2[t + off]; }
    __syncthreads();
  }
  const float mean = s1[0] * (1.0f / DM);
  const float var  = s2[0] * (1.0f / DM) - mean * mean;
  const float rstd = rsqrtf(var + 1e-5f);
  bf16_t* orow = out + (size_t)blockIdx.x * DM;
#pragma unroll
  for (int i = 0; i < 4; ++i) {
    const int col = t + i * 128;
    orow[col] = f2bf((v[i] - mean) * rstd * gam[col] + bet[col]);
  }
}

// Depthwise causal conv (CONV=4) over L + SiLU; reads xin = xz[:, :, 0:DI].
__global__ __launch_bounds__(256) void conv_silu(
    const float* __restrict__ xz, const float* __restrict__ cw,
    const float* __restrict__ cb, float* __restrict__ xcf,
    bf16_t* __restrict__ xcb) {
  const int idx = blockIdx.x * 256 + threadIdx.x;   // < MROWS*DI
  const int c   = idx & (DI - 1);
  const int r   = idx >> 10;                        // b*L + l
  const int l   = r & (SEQL - 1);
  float acc = cb[c];
#pragma unroll
  for (int k = 0; k < 4; ++k) {
    const int li = l + k - 3;
    if (li >= 0) acc += cw[c * 4 + k] * xz[(size_t)(r + k - 3) * (2 * DI) + c];
  }
  const float s = siluf(acc);
  xcf[idx] = s;
  xcb[idx] = f2bf(s);
}

// dtr (first DT_RANK cols of dbl) -> bf16, packed (MROWS x 32)
__global__ __launch_bounds__(256) void cvt_dtr(
    const float* __restrict__ dbl, bf16_t* __restrict__ dtr) {
  const int idx = blockIdx.x * 256 + threadIdx.x;   // < MROWS*32
  const int m = idx >> 5, j = idx & 31;
  dtr[idx] = f2bf(dbl[(size_t)m * 64 + j]);
}

// Selective scan: one thread per (b, channel). Fuses D-skip and SiLU(z)
// gating; emits bf16 ready for out_proj WMMA.
__global__ __launch_bounds__(256) void scan_gate(
    const float* __restrict__ dt, const float* __restrict__ xcf,
    const float* __restrict__ dbl, const float* __restrict__ xz,
    const float* __restrict__ A_log, const float* __restrict__ Dskip,
    bf16_t* __restrict__ ybf) {
  const int tid = blockIdx.x * 256 + threadIdx.x;   // < MROWS
  const int b = tid >> 10, c = tid & (DI - 1);
  float a[DSTATE], h[DSTATE];
#pragma unroll
  for (int s = 0; s < DSTATE; ++s) {
    a[s] = -__expf(A_log[c * DSTATE + s]);
    h[s] = 0.0f;
  }
  const float Dv = Dskip[c];
  for (int l = 0; l < SEQL; ++l) {
    const int r = b * SEQL + l;
    const float dtv = dt[(size_t)r * DI + c];
    const float xv  = xcf[(size_t)r * DI + c];
    const float* Bp = dbl + (size_t)r * 64 + DTRANK;
    const float* Cp = Bp + DSTATE;
    float acc = 0.0f;
#pragma unroll
    for (int s = 0; s < DSTATE; ++s) {
      h[s] = __expf(dtv * a[s]) * h[s] + dtv * Bp[s] * xv;
      acc += h[s] * Cp[s];
    }
    const float z = xz[(size_t)r * (2 * DI) + DI + c];
    ybf[(size_t)r * DI + c] = f2bf((acc + Dv * xv) * siluf(z));
  }
}

// ---------------------------------------------------------------------------

extern "C" void kernel_launch(void* const* d_in, const int* in_sizes, int n_in,
                              void* d_out, int out_size, void* d_ws, size_t ws_size,
                              hipStream_t stream) {
  const float* in_x      = (const float*)d_in[0];
  const float* ln1_g     = (const float*)d_in[1];
  const float* ln1_b     = (const float*)d_in[2];
  const float* in_w      = (const float*)d_in[3];
  const float* conv_w    = (const float*)d_in[4];
  const float* conv_b    = (const float*)d_in[5];
  const float* xproj_w   = (const float*)d_in[6];
  const float* dtproj_w  = (const float*)d_in[7];
  const float* dtproj_b  = (const float*)d_in[8];
  const float* A_log     = (const float*)d_in[9];
  const float* D_skip    = (const float*)d_in[10];
  const float* out_w     = (const float*)d_in[11];
  const float* ln2_g     = (const float*)d_in[12];
  const float* ln2_b     = (const float*)d_in[13];
  const float* mlp_w1    = (const float*)d_in[14];
  const float* mlp_b1    = (const float*)d_in[15];
  const float* mlp_w2    = (const float*)d_in[16];
  const float* mlp_b2    = (const float*)d_in[17];
  float* out = (float*)d_out;

  // Workspace carve-up (all 256B-aligned by construction)
  char* ws = (char*)d_ws;
  size_t off = 0;
  auto carve = [&](size_t bytes) { char* p = ws + off; off += (bytes + 255) & ~(size_t)255; return p; };
  float*  xA    = (float*)carve((size_t)MROWS * DM * 4);
  float*  xB    = (float*)carve((size_t)MROWS * DM * 4);
  bf16_t* lnbf  = (bf16_t*)carve((size_t)MROWS * DM * 2);
  float*  xz    = (float*)carve((size_t)MROWS * 2 * DI * 4);
  float*  xcf   = (float*)carve((size_t)MROWS * DI * 4);
  bf16_t* xcb   = (bf16_t*)carve((size_t)MROWS * DI * 2);
  float*  dbl   = (float*)carve((size_t)MROWS * 64 * 4);
  bf16_t* dtrb  = (bf16_t*)carve((size_t)MROWS * DTRANK * 2);
  float*  dt    = (float*)carve((size_t)MROWS * DI * 4);
  bf16_t* ybf   = (bf16_t*)carve((size_t)MROWS * DI * 2);
  bf16_t* h2bf  = (bf16_t*)carve((size_t)MROWS * 2 * DM * 2);
  (void)ws_size; (void)in_sizes; (void)n_in; (void)out_size;

  (void)hipMemcpyAsync(xA, in_x, (size_t)MROWS * DM * 4, hipMemcpyDeviceToDevice, stream);

  const dim3 blk(128);
  for (int d = 0; d < DEPTH; ++d) {
    const float* l_ln1g = ln1_g    + (size_t)d * DM;
    const float* l_ln1b = ln1_b    + (size_t)d * DM;
    const float* l_inw  = in_w     + (size_t)d * (2 * DI) * DM;
    const float* l_cw   = conv_w   + (size_t)d * DI * 4;
    const float* l_cb   = conv_b   + (size_t)d * DI;
    const float* l_xpw  = xproj_w  + (size_t)d * 64 * DI;
    const float* l_dtw  = dtproj_w + (size_t)d * DI * DTRANK;
    const float* l_dtb  = dtproj_b + (size_t)d * DI;
    const float* l_Alog = A_log    + (size_t)d * DI * DSTATE;
    const float* l_Dsk  = D_skip   + (size_t)d * DI;
    const float* l_ow   = out_w    + (size_t)d * DM * DI;
    const float* l_ln2g = ln2_g    + (size_t)d * DM;
    const float* l_ln2b = ln2_b    + (size_t)d * DM;
    const float* l_w1   = mlp_w1   + (size_t)d * (2 * DM) * DM;
    const float* l_b1   = mlp_b1   + (size_t)d * (2 * DM);
    const float* l_w2   = mlp_w2   + (size_t)d * DM * (2 * DM);
    const float* l_b2   = mlp_b2   + (size_t)d * DM;

    // 1) LN1 -> bf16
    ln_to_bf16<<<MROWS, 128, 0, stream>>>(xA, l_ln1g, l_ln1b, lnbf);
    // 2) in_proj: (4096x512) @ (2048x512)^T -> xz (4096x2048)
    gemm_bf16_wmma<0><<<dim3((2 * DI) / 64, MROWS / 128), blk, 0, stream>>>(
        lnbf, l_inw, xz, nullptr, nullptr, nullptr, 2 * DI, DM);
    // 3) causal conv + SiLU
    conv_silu<<<(MROWS * DI) / 256, 256, 0, stream>>>(xz, l_cw, l_cb, xcf, xcb);
    // 4) xproj: (4096x1024) @ (64x1024)^T -> dbl (4096x64)
    gemm_bf16_wmma<0><<<dim3(1, MROWS / 128), blk, 0, stream>>>(
        xcb, l_xpw, dbl, nullptr, nullptr, nullptr, 64, DI);
    // 5) dtr -> bf16
    cvt_dtr<<<(MROWS * DTRANK) / 256, 256, 0, stream>>>(dbl, dtrb);
    // 6) dtproj + bias + softplus: (4096x32) @ (1024x32)^T -> dt (4096x1024)
    gemm_bf16_wmma<EPI_BIAS | EPI_SOFTPLUS>
        <<<dim3(DI / 64, MROWS / 128), blk, 0, stream>>>(
            dtrb, l_dtw, dt, nullptr, l_dtb, nullptr, DI, DTRANK);
    // 7) selective scan + D-skip + SiLU(z) gating -> y (bf16)
    scan_gate<<<MROWS / 256, 256, 0, stream>>>(dt, xcf, dbl, xz, l_Alog, l_Dsk, ybf);
    // 8) out_proj + residual: xB = y @ out_w^T + xA
    gemm_bf16_wmma<EPI_RESID><<<dim3(DM / 64, MROWS / 128), blk, 0, stream>>>(
        ybf, l_ow, xB, nullptr, nullptr, xA, DM, DI);
    // 9) LN2 -> bf16
    ln_to_bf16<<<MROWS, 128, 0, stream>>>(xB, l_ln2g, l_ln2b, lnbf);
    // 10) MLP up + bias + GELU -> bf16 hidden
    gemm_bf16_wmma<EPI_BIAS | EPI_GELU | EPI_BF16OUT>
        <<<dim3((2 * DM) / 64, MROWS / 128), blk, 0, stream>>>(
            lnbf, l_w1, nullptr, h2bf, l_b1, nullptr, 2 * DM, DM);
    // 11) MLP down + bias + residual -> next x (or final output)
    float* dst = (d == DEPTH - 1) ? out : xA;
    gemm_bf16_wmma<EPI_BIAS | EPI_RESID>
        <<<dim3(DM / 64, MROWS / 128), blk, 0, stream>>>(
            h2bf, l_w2, dst, nullptr, l_b2, xB, DM, 2 * DM);
  }
}
